// GNNPolicy_38397007626251
// MI455X (gfx1250) — matrix operand order repaired and checked by reading, not compiled, not run on hardware
//
#include <hip/hip_runtime.h>
#include <hip/hip_bf16.h>

typedef __attribute__((ext_vector_type(16))) __bf16 v16bf;
typedef __attribute__((ext_vector_type(8)))  float  v8f;
typedef __attribute__((ext_vector_type(8)))  unsigned short us8;
typedef unsigned short u16;

#define EMB 128
#define NVAR 14
#define COLS 1038            // NVAR + NC
#define NROWS 32768          // B*NV = 32*1024
#define KANK 1152            // 128 act + 128*8 spline features

static __device__ __forceinline__ u16 f2bf(float f) {
    unsigned u = __float_as_uint(f);
    unsigned r = (u + 0x7FFFu + ((u >> 16) & 1u)) >> 16;   // RNE
    return (u16)r;
}
static __device__ __forceinline__ float bf2f(u16 h) {
    return __uint_as_float(((unsigned)h) << 16);
}

// ---------------------------------------------------------------------------
// Embed: x = obs[:, :14] @ W_emb^T   (K=14, pure VALU)
// Emits x row-major f32 (residual path) and x^T bf16 [b][e][v] (GEMM B-op)
// ---------------------------------------------------------------------------
__global__ __launch_bounds__(128)
void embed_kernel(const float* __restrict__ obs, const float* __restrict__ Wemb,
                  float* __restrict__ xF, u16* __restrict__ xTBf) {
    int row = blockIdx.x, e = threadIdx.x;
    const float* o = obs + (long long)row * COLS;
    const float* w = Wemb + e * NVAR;
    float s = 0.f;
#pragma unroll
    for (int f = 0; f < NVAR; ++f) s += o[f] * w[f];
    xF[(long long)row * EMB + e] = s;
    long long b = row >> 10, v = row & 1023;
    xTBf[(b << 17) + ((long long)e << 10) + v] = f2bf(s);
}

// ---------------------------------------------------------------------------
// Convert adjacency slab to bf16, both row-major and transposed (per batch)
// ---------------------------------------------------------------------------
__global__ __launch_bounds__(256)
void conv_a_kernel(const float* __restrict__ obs, u16* __restrict__ aBf,
                   u16* __restrict__ aTBf) {
    long long idx = (long long)blockIdx.x * 256 + threadIdx.x; // < 32*1024*1024
    long long b  = idx >> 20;
    int vc = (int)(idx & 0xFFFFF);
    int v = vc >> 10, c = vc & 1023;
    float val = obs[b * 1024LL * COLS + (long long)v * COLS + NVAR + c];
    u16 h = f2bf(val);
    aBf [(b << 20) + ((long long)v << 10) + c] = h;
    aTBf[(b << 20) + ((long long)c << 10) + v] = h;
}

// ---------------------------------------------------------------------------
// Prep: lin weights. For C = X @ W^T, B = W^T (KxN) so B^T (N-major) = W
// itself -> plain f32->bf16 cast of the 3 weight matrices.
// ---------------------------------------------------------------------------
__global__ __launch_bounds__(256)
void prep_lin(const float* __restrict__ w1, const float* __restrict__ w2,
              const float* __restrict__ w3, u16* __restrict__ wt) {
    int idx = blockIdx.x * 256 + threadIdx.x;  // < 3*128*128
    if (idx >= 3 * EMB * EMB) return;
    int l = idx >> 14, r = idx & 16383;
    const float* w = (l == 0) ? w1 : ((l == 1) ? w2 : w3);
    wt[idx] = f2bf(w[r]);
}

// Combined KAN weight, stored transposed (o x 1152): wt[o][k]
__global__ __launch_bounds__(256)
void prep_kan_bf(const float* __restrict__ base, const float* __restrict__ spline,
                 const float* __restrict__ scaler, u16* __restrict__ wt) {
    int idx = blockIdx.x * 256 + threadIdx.x;  // < 128*1152
    if (idx >= EMB * KANK) return;
    int o = idx / KANK, k = idx % KANK;
    float v;
    if (k < EMB) v = base[o * EMB + k];
    else { int r = k - EMB, i = r >> 3, j = r & 7;
           v = spline[(o * EMB + i) * 8 + j] * scaler[o * EMB + i]; }
    wt[idx] = f2bf(v);
}

__global__ __launch_bounds__(256)
void prep_kan3(const float* __restrict__ base, const float* __restrict__ spline,
               const float* __restrict__ scaler, float* __restrict__ w) {
    int k = blockIdx.x * 256 + threadIdx.x;
    if (k >= KANK) return;
    float v;
    if (k < EMB) v = base[k];
    else { int r = k - EMB, i = r >> 3, j = r & 7; v = spline[i * 8 + j] * scaler[i]; }
    w[k] = v;
}

// ---------------------------------------------------------------------------
// Tiled WMMA GEMM: C(M x 128) = A(M x K, bf16, lda) @ B(K x 128)
// B supplied TRANSPOSED (N-major: BT[n][k], row length ldbT).
// 128 thr = 4 waves; wave -> 32 x 128 strip: 2 A fragments x 8 B fragments
// = 16 wmma per K-step of 32; each B fragment (2x ds_load_b128) feeds 2 wmma.
// Staging: thread t copies BT row n=t, 32 contiguous bf16 (4x b128 ld/st).
// Optional bf16 output written transposed per 1024-row batch: [m>>10][n][m&1023]
// ---------------------------------------------------------------------------
#define BSTR 40   // LDS column stride (bf16 units); 80B, 16B-aligned
__global__ __launch_bounds__(128)
void gemm_bf16_n128(const u16* __restrict__ A, const u16* __restrict__ BT,
                    float* __restrict__ Cf, u16* __restrict__ CbfT,
                    const float* __restrict__ bias,
                    int K, int lda, int ldbT,
                    long long sA, long long sBT, long long sC) {
    __shared__ __align__(16) u16 ldsB[EMB * BSTR];
    const int bm = blockIdx.x, bz = blockIdx.y;
    const u16* Ab  = A  + (long long)bz * sA;
    const u16* BTb = BT + (long long)bz * sBT;
    const int tid = threadIdx.x;
    const int wave = tid >> 5, lane = tid & 31;
    const int half = lane >> 4, lrow = lane & 15;
    const int row0 = bm * 128 + wave * 32;     // 32 rows per wave

    v8f acc[2][8];
    const v8f vz = {0.f, 0.f, 0.f, 0.f, 0.f, 0.f, 0.f, 0.f};
#pragma unroll
    for (int t = 0; t < 2; ++t)
#pragma unroll
        for (int j = 0; j < 8; ++j) acc[t][j] = vz;

    for (int k0 = 0; k0 < K; k0 += 32) {
        __syncthreads();
        // stage B tile (128 cols x 32 k) : thread t owns column n = t
        {
            const us8* src = (const us8*)(BTb + (long long)tid * ldbT + k0);
            us8* dst = (us8*)&ldsB[tid * BSTR];
#pragma unroll
            for (int i = 0; i < 4; ++i) dst[i] = src[i];
        }
        __syncthreads();

        // A fragments: ISA 16-bit A layout (lanes 0-15: K 0-7 / 16-23;
        //              lanes 16-31: K 8-15 / 24-31), row M = lane%16
        union { v16bf v; us8 h[2]; } af[2];
#pragma unroll
        for (int t = 0; t < 2; ++t) {
            const u16* arow = Ab + (long long)(row0 + t * 16 + lrow) * lda + k0;
            af[t].h[0] = *(const us8*)(arow + half * 8);
            af[t].h[1] = *(const us8*)(arow + 16 + half * 8);
        }

#pragma unroll
        for (int j = 0; j < 8; ++j) {
            // B fragment: lane -> N = j*16 + lane%16; elems e<8 -> K = 8*half + e, e>=8 -> K = 16 + 8*half + (e-8)
            union { v16bf v; us8 h[2]; } bfm;
            const u16* bcol = &ldsB[(j * 16 + lrow) * BSTR + half * 8];
            bfm.h[0] = *(const us8*)(bcol);
            bfm.h[1] = *(const us8*)(bcol + 16);
            acc[0][j] = __builtin_amdgcn_wmma_f32_16x16x32_bf16(
                false, af[0].v, false, bfm.v, (short)0, acc[0][j], false, false);
            acc[1][j] = __builtin_amdgcn_wmma_f32_16x16x32_bf16(
                false, af[1].v, false, bfm.v, (short)0, acc[1][j], false, false);
        }
    }

    // C layout: lanes 0-15 -> N=lane, M=r; lanes 16-31 -> N=lane-16, M=r+8
#pragma unroll
    for (int t = 0; t < 2; ++t) {
#pragma unroll
        for (int j = 0; j < 8; ++j) {
            int n = j * 16 + lrow;
            float bv = bias ? bias[n] : 0.f;
#pragma unroll
            for (int r = 0; r < 8; ++r) {
                int m = row0 + t * 16 + r + half * 8;
                float v = acc[t][j][r] + bv;
                if (Cf) Cf[(long long)bz * sC + (long long)m * EMB + n] = v;
                if (CbfT) {
                    long long b = m >> 10, vv = m & 1023;
                    CbfT[(b << 17) + ((long long)n << 10) + vv] = f2bf(v);
                }
            }
        }
    }
}

// ---------------------------------------------------------------------------
// Fused LayerNorm -> tanh -> (+residual), emit f32 and/or bf16 (row-major)
// ---------------------------------------------------------------------------
__global__ __launch_bounds__(128)
void ln_tanh_kernel(const float* __restrict__ in, const float* __restrict__ g,
                    const float* __restrict__ b, const float* __restrict__ add,
                    float* __restrict__ outF, u16* __restrict__ outBf) {
    int row = blockIdx.x, e = threadIdx.x;
    __shared__ float red[128];
    long long idx = (long long)row * EMB + e;
    float v = in[idx];
    red[e] = v; __syncthreads();
    for (int s = 64; s > 0; s >>= 1) { if (e < s) red[e] += red[e + s]; __syncthreads(); }
    float mu = red[0] * (1.f / 128.f); __syncthreads();
    float d = v - mu;
    red[e] = d * d; __syncthreads();
    for (int s = 64; s > 0; s >>= 1) { if (e < s) red[e] += red[e + s]; __syncthreads(); }
    float var = red[0] * (1.f / 128.f);
    float y = tanhf(d * rsqrtf(var + 1e-5f) * g[e] + b[e]);
    if (add) y += add[idx];
    if (outF)  outF[idx]  = y;
    if (outBf) outBf[idx] = f2bf(y);
}

// ---------------------------------------------------------------------------
// KAN features: [act(x) (128) | b-spline bases (128*8)] -> bf16 (row-major A)
// ---------------------------------------------------------------------------
static __device__ __forceinline__ void bspline8(float x, float* out) {
    float gk[12];
#pragma unroll
    for (int t = 0; t < 12; ++t) gk[t] = (float)(t - 3) * 0.4f - 1.0f;
    float bb[11];
#pragma unroll
    for (int j = 0; j < 11; ++j) bb[j] = (x >= gk[j] && x < gk[j + 1]) ? 1.f : 0.f;
#pragma unroll
    for (int k = 1; k <= 3; ++k) {
#pragma unroll
        for (int j = 0; j < 10; ++j) {
            if (j < 11 - k) {
                float d1 = gk[j + k]     - gk[j];
                float d2 = gk[j + k + 1] - gk[j + 1];
                bb[j] = (x - gk[j]) / d1 * bb[j] + (gk[j + k + 1] - x) / d2 * bb[j + 1];
            }
        }
    }
#pragma unroll
    for (int j = 0; j < 8; ++j) out[j] = bb[j];
}

__global__ __launch_bounds__(128)
void kan_feat_kernel(const float* __restrict__ in, u16* __restrict__ feat,
                     int actType) {   // 0=silu, 1=sigmoid
    int row = blockIdx.x, i = threadIdx.x;
    float x = in[(long long)row * EMB + i];
    float sg = 1.f / (1.f + __expf(-x));
    float a = actType ? sg : x * sg;
    u16* fr = feat + (long long)row * KANK;
    fr[i] = f2bf(a);
    float bs[8];
    bspline8(x, bs);
#pragma unroll
    for (int j = 0; j < 8; ++j) fr[EMB + i * 8 + j] = f2bf(bs[j]);
}

// KAN3 (o=1): per-row dot over 1152 features, then out = y*0.6 + 0.2
__global__ __launch_bounds__(128)
void kan3_kernel(const u16* __restrict__ feat, const float* __restrict__ w,
                 float* __restrict__ out) {
    int row = blockIdx.x, t = threadIdx.x;
    const u16* fr = feat + (long long)row * KANK;
    float s = 0.f;
    for (int k = t; k < KANK; k += 128) s += bf2f(fr[k]) * w[k];
    __shared__ float red[128];
    red[t] = s; __syncthreads();
    for (int st = 64; st > 0; st >>= 1) { if (t < st) red[t] += red[t + st]; __syncthreads(); }
    if (t == 0) out[row] = red[0] * 0.6f + 0.2f;
}

// ---------------------------------------------------------------------------
extern "C" void kernel_launch(void* const* d_in, const int* in_sizes, int n_in,
                              void* d_out, int out_size, void* d_ws, size_t ws_size,
                              hipStream_t stream) {
    const float* obs  = (const float*)d_in[0];
    const float* Wemb = (const float*)d_in[1];
    const float* lng[4] = {(const float*)d_in[2], (const float*)d_in[4],
                           (const float*)d_in[6], (const float*)d_in[8]};
    const float* lnb[4] = {(const float*)d_in[3], (const float*)d_in[5],
                           (const float*)d_in[7], (const float*)d_in[9]};
    const float* linw[3] = {(const float*)d_in[10], (const float*)d_in[12], (const float*)d_in[14]};
    const float* linb[3] = {(const float*)d_in[11], (const float*)d_in[13], (const float*)d_in[15]};
    const float* kbase[3] = {(const float*)d_in[16], (const float*)d_in[19], (const float*)d_in[22]};
    const float* kspl[3]  = {(const float*)d_in[17], (const float*)d_in[20], (const float*)d_in[23]};
    const float* kscal[3] = {(const float*)d_in[18], (const float*)d_in[21], (const float*)d_in[24]};
    float* out = (float*)d_out;

    char* ws = (char*)d_ws;
    size_t off = 0;
    auto alloc = [&](size_t bytes) -> void* {
        void* p = ws + off; off += (bytes + 255) & ~(size_t)255; return p;
    };
    u16*  aBf    = (u16*) alloc(67108864);              // 32x1024x1024 bf16
    u16*  aTBf   = (u16*) alloc(67108864);
    float* xF    = (float*)alloc(16777216);             // 32768x128 f32
    u16*  xTBf   = (u16*) alloc(8388608);               // [b][e][v] bf16
    float* g1F   = (float*)alloc(16777216);             // einsum outputs (reused)
    u16*  linBfT = (u16*) alloc(8388608);               // lin outputs, [b][n][v]
    float* xcCF  = (float*)alloc(16777216);
    u16*  xcCBf  = (u16*) alloc(8388608);
    float* xcvF  = (float*)alloc(16777216);
    u16*  xcvBf  = (u16*) alloc(8388608);
    u16*  t7Bf   = (u16*) alloc(8388608);
    float* x2F   = (float*)alloc(16777216);
    u16*  wtLin  = (u16*) alloc((size_t)3 * EMB * EMB * 2);
    u16*  kanW1  = (u16*) alloc((size_t)EMB * KANK * 2);
    u16*  kanW2  = (u16*) alloc((size_t)EMB * KANK * 2);
    float* kanW3 = (float*)alloc(KANK * 4);
    u16*  feat   = (u16*) alloc((size_t)NROWS * KANK * 2);
    float* mF    = (float*)alloc(16777216);

    const long long S_A  = 1LL << 20;   // per-batch adjacency stride (elems)
    const long long S_BT = 1LL << 17;   // per-batch x^T / lin^T stride
    const long long S_C  = 1024LL * EMB;

    // --- prep ---
    embed_kernel <<<NROWS, 128, 0, stream>>>(obs, Wemb, xF, xTBf);
    conv_a_kernel<<<131072, 256, 0, stream>>>(obs, aBf, aTBf);
    prep_lin     <<<192, 256, 0, stream>>>(linw[0], linw[1], linw[2], wtLin);
    prep_kan_bf  <<<576, 256, 0, stream>>>(kbase[0], kspl[0], kscal[0], kanW1);
    prep_kan_bf  <<<576, 256, 0, stream>>>(kbase[1], kspl[1], kscal[1], kanW2);
    prep_kan3    <<<5, 256, 0, stream>>>(kbase[2], kspl[2], kscal[2], kanW3);

    dim3 gB(8, 32), gL(256, 1);

    // t1 = a^T @ x ; xc_c = tanh(LN1(t1))
    gemm_bf16_n128<<<gB, 128, 0, stream>>>(aTBf, xTBf, g1F, nullptr, nullptr,
                                           1024, 1024, 1024, S_A, S_BT, S_C);
    ln_tanh_kernel<<<NROWS, 128, 0, stream>>>(g1F, lng[0], lnb[0], nullptr, xcCF, xcCBf);
    // t2 = xc_c @ lin1^T + b1  (bf16 output stored transposed per batch)
    gemm_bf16_n128<<<gL, 128, 0, stream>>>(xcCBf, wtLin, nullptr, linBfT, linb[0],
                                           EMB, EMB, EMB, 0, 0, 0);
    // t3 = a @ t2 ; x_c_v = tanh(LN2(t3)) + x
    gemm_bf16_n128<<<gB, 128, 0, stream>>>(aBf, linBfT, g1F, nullptr, nullptr,
                                           1024, 1024, 1024, S_A, S_BT, S_C);
    ln_tanh_kernel<<<NROWS, 128, 0, stream>>>(g1F, lng[1], lnb[1], xF, xcvF, xcvBf);
    // t5 = x_c_v @ lin2^T + b2
    gemm_bf16_n128<<<gL, 128, 0, stream>>>(xcvBf, wtLin + EMB * EMB, nullptr, linBfT,
                                           linb[1], EMB, EMB, EMB, 0, 0, 0);
    // t6 = a^T @ t5 ; t7 = tanh(LN3(t6)) + xc_c
    gemm_bf16_n128<<<gB, 128, 0, stream>>>(aTBf, linBfT, g1F, nullptr, nullptr,
                                           1024, 1024, 1024, S_A, S_BT, S_C);
    ln_tanh_kernel<<<NROWS, 128, 0, stream>>>(g1F, lng[2], lnb[2], xcCF, nullptr, t7Bf);
    // t8 = t7 @ lin3^T + b3
    gemm_bf16_n128<<<gL, 128, 0, stream>>>(t7Bf, wtLin + 2 * EMB * EMB, nullptr, linBfT,
                                           linb[2], EMB, EMB, EMB, 0, 0, 0);
    // t9 = a @ t8 ; x2 = tanh(LN4(t9)) + x_c_v
    gemm_bf16_n128<<<gB, 128, 0, stream>>>(aBf, linBfT, g1F, nullptr, nullptr,
                                           1024, 1024, 1024, S_A, S_BT, S_C);
    ln_tanh_kernel<<<NROWS, 128, 0, stream>>>(g1F, lng[3], lnb[3], xcvF, x2F, nullptr);

    // KAN1: silu features + spline bases -> GEMM K=1152
    kan_feat_kernel<<<NROWS, 128, 0, stream>>>(x2F, feat, 0);
    gemm_bf16_n128<<<gL, 128, 0, stream>>>(feat, kanW1, mF, nullptr, nullptr,
                                           KANK, KANK, KANK, 0, 0, 0);
    // KAN2
    kan_feat_kernel<<<NROWS, 128, 0, stream>>>(mF, feat, 0);
    gemm_bf16_n128<<<gL, 128, 0, stream>>>(feat, kanW2, mF, nullptr, nullptr,
                                           KANK, KANK, KANK, 0, 0, 0);
    // KAN3 (sigmoid features, o=1 dot) + final affine
    kan_feat_kernel<<<NROWS, 128, 0, stream>>>(mF, feat, 1);
    kan3_kernel<<<NROWS, 128, 0, stream>>>(feat, kanW3, out);
}
